// STB_40458591929166
// MI455X (gfx1250) — matrix operand order repaired and checked by reading, not compiled, not run on hardware
//
#include <hip/hip_runtime.h>
#include <cstdint>

// ---------------- problem constants ----------------
#define BATCH 2
#define SEQ   8192
#define EMB   1024
#define HEADS 16
#define HDIM  64
#define CHUNK 128
#define NCHUNK (SEQ / CHUNK)          // 64
#define AH    (HEADS * HDIM)          // 1024
#define MTOT  (BATCH * SEQ)           // 16384
#define CLAMPV 4.5951198501345898f    // log(0.99/0.01)

// ---------------- WMMA vector types ----------------
typedef __attribute__((ext_vector_type(16))) _Float16 v16h;
typedef __attribute__((ext_vector_type(8)))  _Float16 v8h;   // 16 bytes
typedef __attribute__((ext_vector_type(4)))  _Float16 v4h;   // 8  bytes
typedef __attribute__((ext_vector_type(8)))  float    v8f;

union FragH { v16h v; v8h h[2]; };    // ext_vector members -> trivially copyable union

// ---------------- CDNA5 async global->LDS copy (ASYNCcnt path) --------
// ISA 15.18.3 op 98: GLOBAL_LOAD_ASYNC_TO_LDS_B128. VDST = per-lane LDS byte
// offset; VADDR = 64-bit global address. Generic pointers to __shared__ carry
// the LDS offset in their low 32 bits.
__device__ __forceinline__ void async_copy_b128(_Float16* lds_dst, const _Float16* gsrc) {
    unsigned lds_off = (unsigned)(uintptr_t)lds_dst;
    asm volatile("global_load_async_to_lds_b128 %0, %1, off"
                 :: "v"(lds_off), "v"(gsrc)
                 : "memory");
}
__device__ __forceinline__ void wait_async0() {
    asm volatile("s_wait_asynccnt 0x0" ::: "memory");
}

// =====================================================================
// 1) f32 -> f16 straight convert (x), 4 elems/thread, vectorized
// =====================================================================
__global__ __launch_bounds__(256)
void conv_f32_f16(const float* __restrict__ in, _Float16* __restrict__ out, long n4) {
    long i = (long)blockIdx.x * 256 + threadIdx.x;
    if (i >= n4) return;
    float4 f = ((const float4*)in)[i];
    v4h o;
    o.x = (_Float16)f.x; o.y = (_Float16)f.y; o.z = (_Float16)f.z; o.w = (_Float16)f.w;
    ((v4h*)out)[i] = o;
}

// =====================================================================
// 2) (K,N) f32 -> (N,K) f16 tiled transpose-convert for weights
// =====================================================================
__global__ __launch_bounds__(256)
void transpose_f32_f16(const float* __restrict__ in, _Float16* __restrict__ out, int K, int N) {
    __shared__ float t[32][33];
    int x  = blockIdx.x * 32 + threadIdx.x;   // n
    int y0 = blockIdx.y * 32;                 // k base
#pragma unroll
    for (int i = 0; i < 4; ++i) {
        int y = y0 + threadIdx.y + i * 8;
        if (x < N && y < K) t[threadIdx.y + i * 8][threadIdx.x] = in[(size_t)y * N + x];
    }
    __syncthreads();
    int k = y0 + threadIdx.x;
#pragma unroll
    for (int i = 0; i < 4; ++i) {
        int n = blockIdx.x * 32 + threadIdx.y + i * 8;
        if (n < N && k < K) out[(size_t)n * K + k] = (_Float16)t[threadIdx.x][threadIdx.y + i * 8];
    }
}

// =====================================================================
// 3) Main WMMA GEMM: C(M,N) = A(M,K) * Bt(N,K)^T + bias
//    128x128 tile / WG, BK=32, 8 waves. Double-buffered LDS with async
//    global->LDS DMA of tile k+1 overlapping WMMA on tile k.
// =====================================================================
#define BM 128
#define BN 128
#define BK 32
#define BKP 40   // padded LDS stride (halves); 80B keeps 16B alignment, spreads banks

template<bool HALF_OUT>
__global__ __launch_bounds__(256)
void gemm_f16(const _Float16* __restrict__ A,   // (M,K)
              const _Float16* __restrict__ Bt,  // (N,K)
              const float*    __restrict__ bias,// (N)
              void*           __restrict__ Cout,// (M,N) f32 or f16
              int M, int N, int K) {
    __shared__ __align__(16) _Float16 As[2][BM * BKP];
    __shared__ __align__(16) _Float16 Bs[2][BN * BKP];

    const int tid  = threadIdx.x;
    const int wave = tid >> 5;
    const int lane = tid & 31;
    const int m0   = blockIdx.y * BM;
    const int n0   = blockIdx.x * BN;

    v8f zero = {};
    v8f acc[8];
#pragma unroll
    for (int i = 0; i < 8; ++i) acc[i] = zero;

    const int lrow = lane & 15;
    const int koff = (lane >> 4) * 8;   // A-frag K sub-offset (ISA 16-bit A layout)
    const int kb   = (lane >> 4) * 16;

    // this thread's two 16B chunks of each tile
    const int row0 = tid >> 1;                 // 0..127  (chunks 0,1 of row)
    const int ch0  = (tid & 1) * 2;            // 0 or 2  -> two consecutive chunks
    // tile copy: 512 chunks per array, 256 threads -> 2 chunks each (adjacent)
    auto load_tile = [&](int buf, int k0) {
#pragma unroll
        for (int i = 0; i < 2; ++i) {
            int ch = ch0 + i;
            async_copy_b128(&As[buf][row0 * BKP + ch * 8],
                            &A [(size_t)(m0 + row0) * K + k0 + ch * 8]);
            async_copy_b128(&Bs[buf][row0 * BKP + ch * 8],
                            &Bt[(size_t)(n0 + row0) * K + k0 + ch * 8]);
        }
    };

    load_tile(0, 0);
    wait_async0();
    __syncthreads();

    const int NIT = K / BK;
    for (int it = 0; it < NIT; ++it) {
        const int cur = it & 1;
        if (it + 1 < NIT) load_tile(1 - cur, (it + 1) * BK);   // DMA next tile

        FragH a;
        a.h[0] = *(const v8h*)&As[cur][(wave * 16 + lrow) * BKP + koff];
        a.h[1] = *(const v8h*)&As[cur][(wave * 16 + lrow) * BKP + 16 + koff];
#pragma unroll
        for (int g = 0; g < 2; ++g) {           // 2 groups of 4 B-frags -> fewer waits
            FragH bf[4];
#pragma unroll
            for (int q = 0; q < 4; ++q) {
                int nt = g * 4 + q;
                bf[q].h[0] = *(const v8h*)&Bs[cur][(nt * 16 + lrow) * BKP + koff];
                bf[q].h[1] = *(const v8h*)&Bs[cur][(nt * 16 + lrow) * BKP + 16 + koff];
            }
#pragma unroll
            for (int q = 0; q < 4; ++q) {
                int nt = g * 4 + q;
                acc[nt] = __builtin_amdgcn_wmma_f32_16x16x32_f16(
                    false, a.v, false, bf[q].v, (short)0, acc[nt], false, false);
            }
        }
        wait_async0();       // next tile resident
        __syncthreads();     // all waves done reading cur / writing next
    }

#pragma unroll
    for (int nt = 0; nt < 8; ++nt) {
        int col = n0 + nt * 16 + lrow;
        float bs = bias[col];
#pragma unroll
        for (int r = 0; r < 8; ++r) {
            int row = m0 + wave * 16 + r + ((lane >> 4) << 3);   // C-frag M mapping
            float val = acc[nt][r] + bs;
            if (HALF_OUT) ((_Float16*)Cout)[(size_t)row * N + col] = (_Float16)val;
            else          ((float*)   Cout)[(size_t)row * N + col] = val;
        }
    }
}

// =====================================================================
// 4) Gate GEMM: score_t(B,H,L) = clamp(x @ W_gate + b, CLAMP), N = 16
// =====================================================================
__global__ __launch_bounds__(256)
void gemm_score(const _Float16* __restrict__ A,    // (MTOT, EMB)
                const _Float16* __restrict__ WgT,  // (HEADS, EMB)
                const float*    __restrict__ b_gate,
                float*          __restrict__ score_t) {
    const int tid  = threadIdx.x;
    const int wave = tid >> 5;
    const int lane = tid & 31;
    const int m0   = (blockIdx.x * 8 + wave) * 16;
    const int lrow = lane & 15;
    const int koff = (lane >> 4) * 8;
    const int kb   = (lane >> 4) * 16;

    v8f acc = {};
    for (int k0 = 0; k0 < EMB; k0 += 32) {
        FragH a, b;
        a.h[0] = *(const v8h*)&A[(size_t)(m0 + lrow) * EMB + k0 + koff];
        a.h[1] = *(const v8h*)&A[(size_t)(m0 + lrow) * EMB + k0 + 16 + koff];
        b.h[0] = *(const v8h*)&WgT[(size_t)lrow * EMB + k0 + koff];
        b.h[1] = *(const v8h*)&WgT[(size_t)lrow * EMB + k0 + 16 + koff];
        if (k0 + 32 < EMB)
            __builtin_prefetch(&A[(size_t)(m0 + lrow) * EMB + k0 + 32 + koff], 0, 1);
        acc = __builtin_amdgcn_wmma_f32_16x16x32_f16(
            false, a.v, false, b.v, (short)0, acc, false, false);
    }
    const int h  = lrow;                     // head = column
    const float bs = b_gate[h];
#pragma unroll
    for (int r = 0; r < 8; ++r) {
        int row = m0 + r + ((lane >> 4) << 3);
        int bb  = row >> 13;                 // /SEQ
        int l   = row & (SEQ - 1);
        score_t[(size_t)(bb * HEADS + h) * SEQ + l] = fminf(acc[r] + bs, CLAMPV);
    }
}

// =====================================================================
// 5) Intra-chunk scan. block_v[m] = sig(-s_m)*block_v[m-1] + sig(s_m)*v_m
//    (algebraically identical to the reference's log-space form, stable)
// =====================================================================
__global__ __launch_bounds__(128)
void chunk_scan(const float*    __restrict__ score_t,  // (B,H,L)
                const _Float16* __restrict__ v,        // (B*L, AH) f16
                _Float16* __restrict__ block_v,        // (B,H,C,M,D) f16
                float* __restrict__ gates,             // (B,H,L)
                float* __restrict__ bv_last,           // (B,H,C,D)
                float* __restrict__ g_last) {          // (B,H,C)
    const int bc = blockIdx.x;           // bh*NCHUNK + c
    const int bh = bc >> 6;
    const int c  = bc & (NCHUNK - 1);
    const int b  = bh >> 4;
    const int h  = bh & (HEADS - 1);
    const int m  = threadIdx.x;

    __shared__ float ls[CHUNK], sp_s[CHUNK], sn_s[CHUNK];

    float s  = score_t[(size_t)bh * SEQ + c * CHUNK + m];
    float e  = __expf(-fabsf(s));
    float inv = 1.0f / (1.0f + e);
    float sp = (s >= 0.0f) ? inv : e * inv;          // sigmoid(s)
    float sn = (s >= 0.0f) ? e * inv : inv;          // sigmoid(-s)
    float lsn = -(fmaxf(s, 0.0f) + log1pf(e));       // logsig(-s) = -softplus(s)
    sp_s[m] = sp; sn_s[m] = sn; ls[m] = lsn;
    __syncthreads();

    for (int off = 1; off < CHUNK; off <<= 1) {      // Hillis-Steele prefix
        float add = (m >= off) ? ls[m - off] : 0.0f;
        __syncthreads();
        ls[m] += add;
        __syncthreads();
    }
    float g = __expf(ls[m]);
    gates[(size_t)bh * SEQ + c * CHUNK + m] = g;
    if (m == CHUNK - 1) g_last[bc] = g;
    __syncthreads();

    if (m < HDIM) {                                  // one thread per d
        const int d = m;
        const _Float16* vp = v + (size_t)(b * SEQ + c * CHUNK) * AH + h * HDIM + d;
        _Float16* bp = block_v + (size_t)bc * CHUNK * HDIM + d;
        float acc = 0.0f;
#pragma unroll 4
        for (int j = 0; j < CHUNK; ++j) {
            acc = sn_s[j] * acc + sp_s[j] * (float)vp[(size_t)j * AH];
            bp[(size_t)j * HDIM] = (_Float16)acc;
        }
        bv_last[(size_t)bc * HDIM + d] = acc;
    }
}

// =====================================================================
// 6) Cross-chunk carry scan: carry_{c+1} = bv_last[c] + g_last[c]*carry_c
// =====================================================================
__global__ __launch_bounds__(64)
void carry_scan(const float* __restrict__ bv_last, const float* __restrict__ g_last,
                float* __restrict__ carry) {        // (B,H,C,D): carry BEFORE chunk c
    const int bh = blockIdx.x;
    const int d  = threadIdx.x;
    float cur = 0.0f;
    for (int c = 0; c < NCHUNK; ++c) {
        int bc = bh * NCHUNK + c;
        carry[(size_t)bc * HDIM + d] = cur;
        cur = bv_last[(size_t)bc * HDIM + d] + g_last[bc] * cur;
    }
}

// =====================================================================
// 7) Fuse: y = block_v + gates*carry, shift right by 1 along L,
//    multiply by sigmoid(og), emit f16 for the final GEMM
// =====================================================================
__global__ __launch_bounds__(256)
void fuse_y1(const _Float16* __restrict__ block_v, const float* __restrict__ gates,
             const float* __restrict__ carry, const _Float16* __restrict__ og,
             _Float16* __restrict__ y1h) {
    size_t i = (size_t)blockIdx.x * 256 + threadIdx.x;   // over B*L*AH
    int d  = (int)(i & (HDIM - 1));
    int h  = (int)((i >> 6) & (HEADS - 1));
    size_t bl = i >> 10;                                 // b*SEQ + l
    int b  = (int)(bl >> 13);
    int l  = (int)(bl & (SEQ - 1));

    float y = 0.0f;
    if (l > 0) {
        int lp = l - 1;
        int c  = lp >> 7;
        int m  = lp & (CHUNK - 1);
        int bh = b * HEADS + h;
        size_t bc = (size_t)bh * NCHUNK + c;
        float bv = (float)block_v[(bc * CHUNK + m) * HDIM + d];
        float g  = gates[(size_t)bh * SEQ + lp];
        float cr = carry[bc * HDIM + d];
        y = bv + g * cr;
    }
    float o = (float)og[i];
    float y1 = y / (1.0f + __expf(-o));
    y1h[i] = (_Float16)y1;
}

// =====================================================================
// launch
// =====================================================================
extern "C" void kernel_launch(void* const* d_in, const int* in_sizes, int n_in,
                              void* d_out, int out_size, void* d_ws, size_t ws_size,
                              hipStream_t stream) {
    (void)in_sizes; (void)n_in; (void)out_size; (void)ws_size;

    const float* x      = (const float*)d_in[0];
    const float* W_gate = (const float*)d_in[1];
    const float* b_gate = (const float*)d_in[2];
    const float* W_val  = (const float*)d_in[3];
    const float* b_val  = (const float*)d_in[4];
    const float* W_og   = (const float*)d_in[5];
    const float* b_og   = (const float*)d_in[6];
    const float* W_out  = (const float*)d_in[7];
    const float* b_out  = (const float*)d_in[8];
    float* out = (float*)d_out;

    char* w = (char*)d_ws;
    auto alloc = [&](size_t bytes) {
        char* p = w;
        w += (bytes + 255) & ~(size_t)255;
        return p;
    };
    _Float16* xh      = (_Float16*)alloc((size_t)MTOT * EMB * 2);
    _Float16* WvT     = (_Float16*)alloc((size_t)AH * EMB * 2);
    _Float16* WogT    = (_Float16*)alloc((size_t)AH * EMB * 2);
    _Float16* WoutT   = (_Float16*)alloc((size_t)EMB * AH * 2);
    _Float16* WgT     = (_Float16*)alloc((size_t)HEADS * EMB * 2);
    float*    score_t = (float*)   alloc((size_t)BATCH * HEADS * SEQ * 4);
    _Float16* vbuf    = (_Float16*)alloc((size_t)MTOT * AH * 2);
    _Float16* ogbuf   = (_Float16*)alloc((size_t)MTOT * AH * 2);
    _Float16* block_v = (_Float16*)alloc((size_t)BATCH * HEADS * SEQ * HDIM * 2);
    float*    gates   = (float*)   alloc((size_t)BATCH * HEADS * SEQ * 4);
    float*    bv_last = (float*)   alloc((size_t)BATCH * HEADS * NCHUNK * HDIM * 4);
    float*    g_last  = (float*)   alloc((size_t)BATCH * HEADS * NCHUNK * 4);
    float*    carry   = (float*)   alloc((size_t)BATCH * HEADS * NCHUNK * HDIM * 4);
    _Float16* y1h     = (_Float16*)alloc((size_t)MTOT * AH * 2);

    // 1) converts / transposes
    {
        long n4 = (long)MTOT * EMB / 4;
        conv_f32_f16<<<dim3((unsigned)((n4 + 255) / 256)), dim3(256), 0, stream>>>(x, xh, n4);
        dim3 tb(32, 8);
        transpose_f32_f16<<<dim3(AH / 32, EMB / 32), tb, 0, stream>>>(W_val,  WvT,   EMB, AH);
        transpose_f32_f16<<<dim3(AH / 32, EMB / 32), tb, 0, stream>>>(W_og,   WogT,  EMB, AH);
        transpose_f32_f16<<<dim3(EMB / 32, AH / 32), tb, 0, stream>>>(W_out,  WoutT, AH,  EMB);
        transpose_f32_f16<<<dim3(1, EMB / 32),       tb, 0, stream>>>(W_gate, WgT,   EMB, HEADS);
    }

    // 2) score = clamp(x@W_gate + b), written transposed (B,H,L)
    gemm_score<<<dim3(MTOT / 128), dim3(256), 0, stream>>>(xh, WgT, b_gate, score_t);

    // 3) v = x@W_val + b ; og = x@W_og + b  (f16 outputs)
    dim3 ggrid(AH / BN, MTOT / BM);
    gemm_f16<true><<<ggrid, dim3(256), 0, stream>>>(xh, WvT,  b_val, vbuf,  MTOT, AH, EMB);
    gemm_f16<true><<<ggrid, dim3(256), 0, stream>>>(xh, WogT, b_og,  ogbuf, MTOT, AH, EMB);

    // 4) intra-chunk scan
    chunk_scan<<<dim3(BATCH * HEADS * NCHUNK), dim3(CHUNK), 0, stream>>>(
        score_t, vbuf, block_v, gates, bv_last, g_last);

    // 5) cross-chunk carry scan
    carry_scan<<<dim3(BATCH * HEADS), dim3(HDIM), 0, stream>>>(bv_last, g_last, carry);

    // 6) fused gather + shift + og-gate -> y1 (f16)
    fuse_y1<<<dim3((unsigned)((size_t)MTOT * AH / 256)), dim3(256), 0, stream>>>(
        block_v, gates, carry, ogbuf, y1h);

    // 7) out = y1 @ W_out + b_out  (f32 output)
    gemm_f16<false><<<dim3(EMB / BN, MTOT / BM), dim3(256), 0, stream>>>(
        y1h, WoutT, b_out, out, MTOT, EMB, AH);
}
